// DebertaV2Attention_41034117546288
// MI455X (gfx1250) — matrix operand order repaired and checked by reading, not compiled, not run on hardware
//
#include <hip/hip_runtime.h>
#include <hip/hip_bf16.h>

typedef __bf16 bf16_t;
typedef __attribute__((ext_vector_type(16))) __bf16 v16bf;
typedef __attribute__((ext_vector_type(8)))  __bf16 v8bf;
typedef __attribute__((ext_vector_type(8)))  float  v8f;

#define D_MODEL 768
#define NHEAD   12
#define HD      64
#define SEQ     1024
#define BATCH   4
#define SPAN    512
#define WIN     48          // pos-embedding window per 16x32 score tile

// ---------------- fragment helpers (CDNA5 WMMA 16x16x32 bf16 layouts) ----------

static __device__ inline v8f zero8() {
  v8f z;
#pragma unroll
  for (int i = 0; i < 8; ++i) z[i] = 0.f;
  return z;
}

static __device__ inline v16bf combine88(v8bf lo, v8bf hi) {
  v16bf f;
#pragma unroll
  for (int i = 0; i < 8; ++i) { f[i] = lo[i]; f[8 + i] = hi[i]; }
  return f;
}

// A fragment (16xK32): lane(0..15)=row M, lanes 16..31 same rows with K+8 offset.
// Per lane: K = {8h..8h+7, 16+8h..23+8h}  -> two 16B loads.
static __device__ inline v16bf fragA_bf16(const bf16_t* __restrict__ base, int ld,
                                          int lane, int koff) {
  int m = lane & 15, hf = lane >> 4;
  const bf16_t* r = base + m * ld + koff + 8 * hf;
  v8bf lo = *(const v8bf*)(r);
  v8bf hi = *(const v8bf*)(r + 16);
  return combine88(lo, hi);
}

static __device__ inline v16bf fragA_f32(const float* __restrict__ base, int ld, int lane) {
  int m = lane & 15, hf = lane >> 4;
  const float* r = base + m * ld + 8 * hf;
  float4 x0 = *(const float4*)(r);
  float4 x1 = *(const float4*)(r + 4);
  float4 x2 = *(const float4*)(r + 16);
  float4 x3 = *(const float4*)(r + 20);
  v16bf f;
  f[0] = (bf16_t)x0.x;  f[1] = (bf16_t)x0.y;  f[2] = (bf16_t)x0.z;  f[3] = (bf16_t)x0.w;
  f[4] = (bf16_t)x1.x;  f[5] = (bf16_t)x1.y;  f[6] = (bf16_t)x1.z;  f[7] = (bf16_t)x1.w;
  f[8] = (bf16_t)x2.x;  f[9] = (bf16_t)x2.y;  f[10] = (bf16_t)x2.z; f[11] = (bf16_t)x2.w;
  f[12] = (bf16_t)x3.x; f[13] = (bf16_t)x3.y; f[14] = (bf16_t)x3.z; f[15] = (bf16_t)x3.w;
  return f;
}

// B fragment (K32x16): lane(0..15)=col N, per lane K = {8h..8h+7, 16+8h..23+8h}.
static __device__ inline v16bf fragB_bf16(const bf16_t* __restrict__ row, int koff, int hf) {
  const bf16_t* r = row + koff + 8 * hf;
  v8bf lo = *(const v8bf*)(r);
  v8bf hi = *(const v8bf*)(r + 16);
  return combine88(lo, hi);
}

static __device__ inline v16bf fragB_f32(const float* __restrict__ row, int hf) {
  const float* r = row + 8 * hf;
  float4 x0 = *(const float4*)(r);
  float4 x1 = *(const float4*)(r + 4);
  float4 x2 = *(const float4*)(r + 16);
  float4 x3 = *(const float4*)(r + 20);
  v16bf f;
  f[0] = (bf16_t)x0.x;  f[1] = (bf16_t)x0.y;  f[2] = (bf16_t)x0.z;  f[3] = (bf16_t)x0.w;
  f[4] = (bf16_t)x1.x;  f[5] = (bf16_t)x1.y;  f[6] = (bf16_t)x1.z;  f[7] = (bf16_t)x1.w;
  f[8] = (bf16_t)x2.x;  f[9] = (bf16_t)x2.y;  f[10] = (bf16_t)x2.z; f[11] = (bf16_t)x2.w;
  f[12] = (bf16_t)x3.x; f[13] = (bf16_t)x3.y; f[14] = (bf16_t)x3.z; f[15] = (bf16_t)x3.w;
  return f;
}

static __device__ inline v8f wmma_bf16(v16bf a, v16bf b, v8f c) {
  return __builtin_amdgcn_wmma_f32_16x16x32_bf16(false, a, false, b, (short)0, c,
                                                 false, false);
}

static __device__ inline float redmax16(float v) {
  v = fmaxf(v, __shfl_xor(v, 1, 32));
  v = fmaxf(v, __shfl_xor(v, 2, 32));
  v = fmaxf(v, __shfl_xor(v, 4, 32));
  v = fmaxf(v, __shfl_xor(v, 8, 32));
  return v;
}
static __device__ inline float redsum16(float v) {
  v += __shfl_xor(v, 1, 32);
  v += __shfl_xor(v, 2, 32);
  v += __shfl_xor(v, 4, 32);
  v += __shfl_xor(v, 8, 32);
  return v;
}

// ---------------- kernel 1: projection GEMM  out = X @ W^T + b  (bf16 out) -----
// Register-blocked: one wave computes a 32x64 output tile (8 WMMAs / K-step,
// each A frag reused 4x, each B frag reused 2x).
// store_mode 0: row-major [M, 768] bf16.  store_mode 1: V-transposed [B,H,HD,S].
__global__ void __launch_bounds__(128) proj_gemm_kernel(
    const float* __restrict__ X, const float* __restrict__ W,
    const float* __restrict__ bias, bf16_t* __restrict__ out, int M, int store_mode) {
  int wave = blockIdx.x * (blockDim.x >> 5) + (threadIdx.x >> 5);
  int lane = threadIdx.x & 31;
  const int ntn = D_MODEL / 64;                 // 12 N-tiles of 64
  if (wave >= (M / 32) * ntn) return;
  int m0 = (wave / ntn) * 32;
  int n0 = (wave % ntn) * 64;
  int n = lane & 15, hf = lane >> 4;

  const float* A0 = X + (size_t)m0 * D_MODEL;
  const float* A1 = X + (size_t)(m0 + 16) * D_MODEL;
  const float* Br[4];
#pragma unroll
  for (int j = 0; j < 4; ++j) Br[j] = W + (size_t)(n0 + j * 16 + n) * D_MODEL;

  v8f acc[2][4];
#pragma unroll
  for (int mi = 0; mi < 2; ++mi)
#pragma unroll
    for (int j = 0; j < 4; ++j) acc[mi][j] = zero8();

#pragma unroll 2
  for (int k = 0; k < D_MODEL; k += 32) {
    if (k + 32 < D_MODEL) {
      __builtin_prefetch(A0 + (size_t)(lane & 15) * D_MODEL + k + 32, 0, 1);
      __builtin_prefetch(A1 + (size_t)(lane & 15) * D_MODEL + k + 32, 0, 1);
      __builtin_prefetch(Br[0] + k + 32, 0, 1);
    }
    v16bf a0 = fragA_f32(A0 + k, D_MODEL, lane);
    v16bf a1 = fragA_f32(A1 + k, D_MODEL, lane);
#pragma unroll
    for (int j = 0; j < 4; ++j) {
      v16bf b = fragB_f32(Br[j] + k, hf);
      acc[0][j] = wmma_bf16(a0, b, acc[0][j]);
      acc[1][j] = wmma_bf16(a1, b, acc[1][j]);
    }
  }
#pragma unroll
  for (int mi = 0; mi < 2; ++mi) {
#pragma unroll
    for (int j = 0; j < 4; ++j) {
      float bn = bias[n0 + j * 16 + n];
#pragma unroll
      for (int r = 0; r < 8; ++r) {
        int mrow = m0 + mi * 16 + r + 8 * hf;
        float val = acc[mi][j][r] + bn;
        if (store_mode == 0) {
          out[(size_t)mrow * D_MODEL + n0 + j * 16 + n] = (bf16_t)val;
        } else {
          int bb = mrow / SEQ, si = mrow % SEQ;
          int ng = n0 + j * 16 + n, h = ng / HD, d = ng % HD;
          out[((size_t)(bb * NHEAD + h) * HD + d) * SEQ + si] = (bf16_t)val;
        }
      }
    }
  }
}

// ---------------- kernel 2: fused disentangled attention ----------------------
// One wave per (b, h, 16-query tile); online softmax over 32-key tiles.
// 26 WMMAs per tile: 4 QK^T + 6 c2p + 12 p2c + 4 PV, all K-contiguous operands.
__global__ void __launch_bounds__(128) attn_kernel(
    const bf16_t* __restrict__ qb_all, const bf16_t* __restrict__ kb_all,
    const bf16_t* __restrict__ vT_all, const bf16_t* __restrict__ posq,
    const bf16_t* __restrict__ posk, const int* __restrict__ amask,
    bf16_t* __restrict__ ctx_out) {
  __shared__ __align__(16) float  s_c2p[4][16 * WIN];
  __shared__ __align__(16) float  s_p2c[4][32 * WIN];
  __shared__ __align__(16) bf16_t s_p[4][16 * 32];

  int wid = threadIdx.x >> 5;
  int lane = threadIdx.x & 31;
  int wave = blockIdx.x * 4 + wid;
  if (wave >= BATCH * NHEAD * (SEQ / 16)) return;
  int qt = wave % (SEQ / 16);
  int h = (wave / (SEQ / 16)) % NHEAD;
  int b = wave / ((SEQ / 16) * NHEAD);
  int q0 = qt * 16;
  int n = lane & 15, hf = lane >> 4;

  const bf16_t* qbase = qb_all + ((size_t)(b * SEQ + q0)) * D_MODEL + h * HD;
  const bf16_t* kbase = kb_all + (size_t)b * SEQ * D_MODEL + h * HD;
  const bf16_t* vT = vT_all + (size_t)(b * NHEAD + h) * HD * SEQ;
  const bf16_t* pk = posk + h * HD;
  const bf16_t* pq = posq + h * HD;
  const int* mrow = amask + (size_t)b * SEQ * SEQ;  // mask[b,0,:,:]; diag == valid

  // loop-invariant Q fragments (K = 0..31, 32..63)
  v16bf aq0 = fragA_bf16(qbase, D_MODEL, lane, 0);
  v16bf aq1 = fragA_bf16(qbase, D_MODEL, lane, 32);

  float rowmax[8], rowsum[8], vq[8];
  v8f ctx[4];
#pragma unroll
  for (int r = 0; r < 8; ++r) {
    rowmax[r] = -1e30f;
    rowsum[r] = 0.f;
    int qi = q0 + r + 8 * hf;
    vq[r] = mrow[(size_t)qi * SEQ + qi] ? 1.f : 0.f;
  }
#pragma unroll
  for (int s = 0; s < 4; ++s) ctx[s] = zero8();

  const float inv_sqk = 0.07216878364870322f;  // 1/sqrt(3*HD)
  const float inv_sd = 0.125f;                 // 1/sqrt(HD)

  float* c2p = s_c2p[wid];
  float* p2c = s_p2c[wid];
  bf16_t* ptile = s_p[wid];

  for (int k0 = 0; k0 < SEQ; k0 += 32) {
    // ---- content scores: q(16x64) @ k^T -> two 16x16 tiles ----
    v8f sc0 = zero8(), sc1 = zero8();
    {
      const bf16_t* kr0 = kbase + (size_t)(k0 + n) * D_MODEL;
      const bf16_t* kr1 = kbase + (size_t)(k0 + 16 + n) * D_MODEL;
      sc0 = wmma_bf16(aq0, fragB_bf16(kr0, 0, hf), sc0);
      sc0 = wmma_bf16(aq1, fragB_bf16(kr0, 32, hf), sc0);
      sc1 = wmma_bf16(aq0, fragB_bf16(kr1, 0, hf), sc1);
      sc1 = wmma_bf16(aq1, fragB_bf16(kr1, 32, hf), sc1);
    }

    // ---- position window: delta = q - key + SPAN, window base w0 ----
    int w0 = q0 - k0 + SPAN - 31;

    // c2p tile: q @ pos_k[window]^T  -> [16 q][48 win] in LDS
#pragma unroll
    for (int wseg = 0; wseg < 3; ++wseg) {
      int prow = w0 + wseg * 16 + n;
      prow = prow < 0 ? 0 : (prow > 2 * SPAN - 1 ? 2 * SPAN - 1 : prow);
      const bf16_t* pr = pk + (size_t)prow * D_MODEL;
      v8f d = zero8();
      d = wmma_bf16(aq0, fragB_bf16(pr, 0, hf), d);
      d = wmma_bf16(aq1, fragB_bf16(pr, 32, hf), d);
#pragma unroll
      for (int r = 0; r < 8; ++r) c2p[(r + 8 * hf) * WIN + wseg * 16 + n] = d[r];
    }

    // p2c tile: k @ pos_q[window]^T -> [32 key][48 win] in LDS
#pragma unroll
    for (int kh = 0; kh < 2; ++kh) {
      const bf16_t* kb2 = kbase + (size_t)(k0 + kh * 16) * D_MODEL;
      v16bf ak0 = fragA_bf16(kb2, D_MODEL, lane, 0);
      v16bf ak1 = fragA_bf16(kb2, D_MODEL, lane, 32);
#pragma unroll
      for (int wseg = 0; wseg < 3; ++wseg) {
        int prow = w0 + wseg * 16 + n;
        prow = prow < 0 ? 0 : (prow > 2 * SPAN - 1 ? 2 * SPAN - 1 : prow);
        const bf16_t* pr = pq + (size_t)prow * D_MODEL;
        v8f d = zero8();
        d = wmma_bf16(ak0, fragB_bf16(pr, 0, hf), d);
        d = wmma_bf16(ak1, fragB_bf16(pr, 32, hf), d);
#pragma unroll
        for (int r = 0; r < 8; ++r)
          p2c[(kh * 16 + r + 8 * hf) * WIN + wseg * 16 + n] = d[r];
      }
    }
    // same-wave LDS RAW: DS ops execute in program order per wave; each wave
    // owns a private LDS region, so no workgroup barrier is needed.

    // ---- assemble scores + mask + online softmax ----
    float sv[2][8];
    float vk0 = mrow[(size_t)(k0 + n) * SEQ + (k0 + n)] ? 1.f : 0.f;
    float vk1 = mrow[(size_t)(k0 + 16 + n) * SEQ + (k0 + 16 + n)] ? 1.f : 0.f;
#pragma unroll
    for (int t = 0; t < 2; ++t) {
      int kl = n + 16 * t;
      float vk = t ? vk1 : vk0;
#pragma unroll
      for (int r = 0; r < 8; ++r) {
        int m_ = r + 8 * hf;
        int rw = m_ - kl + 31;  // 0..46
        float base = (t ? sc1[r] : sc0[r]);
        float s = base * inv_sqk + c2p[m_ * WIN + rw] * inv_sqk +
                  p2c[kl * WIN + rw] * inv_sd;
        bool ok = (vk > 0.5f) && (vq[r] > 0.5f);
        sv[t][r] = ok ? s : -1e30f;
      }
    }
#pragma unroll
    for (int r = 0; r < 8; ++r) {
      int m_ = r + 8 * hf;
      float tm = fmaxf(sv[0][r], sv[1][r]);
      tm = redmax16(tm);
      float nm = fmaxf(rowmax[r], tm);
      float corr = __expf(rowmax[r] - nm);
      float p0 = sv[0][r] > -1e29f ? __expf(sv[0][r] - nm) : 0.f;
      float p1 = sv[1][r] > -1e29f ? __expf(sv[1][r] - nm) : 0.f;
      float ps = redsum16(p0 + p1);
      rowsum[r] = rowsum[r] * corr + ps;
      rowmax[r] = nm;
#pragma unroll
      for (int s = 0; s < 4; ++s) ctx[s][r] *= corr;
      ptile[m_ * 32 + n] = (bf16_t)p0;
      ptile[m_ * 32 + n + 16] = (bf16_t)p1;
    }

    // ---- P(16x32) @ V(32x64) using transposed V (K-contiguous B frags) ----
    v16bf ap = fragA_bf16(ptile, 32, lane, 0);
#pragma unroll
    for (int s = 0; s < 4; ++s) {
      const bf16_t* vrow = vT + (size_t)(s * 16 + n) * SEQ + k0;
      ctx[s] = wmma_bf16(ap, fragB_bf16(vrow, 0, hf), ctx[s]);
    }
  }

  // ---- epilogue: normalize + store bf16 context [B,S,D] ----
#pragma unroll
  for (int r = 0; r < 8; ++r) {
    float inv = rowsum[r] > 0.f ? 1.f / rowsum[r] : 0.f;
    int tok = q0 + r + 8 * hf;
    size_t base = ((size_t)(b * SEQ + tok)) * D_MODEL + h * HD;
#pragma unroll
    for (int s = 0; s < 4; ++s)
      ctx_out[base + s * 16 + n] = (bf16_t)(ctx[s][r] * inv);
  }
}

// ---------------- kernel 3a: output GEMM + residual (32x64 blocked) -----------
__global__ void __launch_bounds__(128) out_gemm_kernel(
    const bf16_t* __restrict__ ctxb, const float* __restrict__ Wo,
    const float* __restrict__ bo, const float* __restrict__ hidden,
    float* __restrict__ y) {
  int wave = blockIdx.x * (blockDim.x >> 5) + (threadIdx.x >> 5);
  int lane = threadIdx.x & 31;
  const int ntn = D_MODEL / 64;
  if (wave >= (BATCH * SEQ / 32) * ntn) return;
  int m0 = (wave / ntn) * 32;
  int n0 = (wave % ntn) * 64;
  int n = lane & 15, hf = lane >> 4;

  const bf16_t* A0 = ctxb + (size_t)m0 * D_MODEL;
  const bf16_t* A1 = ctxb + (size_t)(m0 + 16) * D_MODEL;
  const float* Br[4];
#pragma unroll
  for (int j = 0; j < 4; ++j) Br[j] = Wo + (size_t)(n0 + j * 16 + n) * D_MODEL;

  v8f acc[2][4];
#pragma unroll
  for (int mi = 0; mi < 2; ++mi)
#pragma unroll
    for (int j = 0; j < 4; ++j) acc[mi][j] = zero8();

#pragma unroll 2
  for (int k = 0; k < D_MODEL; k += 32) {
    if (k + 32 < D_MODEL) __builtin_prefetch(Br[0] + k + 32, 0, 1);
    v16bf a0 = fragA_bf16(A0, D_MODEL, lane, k);
    v16bf a1 = fragA_bf16(A1, D_MODEL, lane, k);
#pragma unroll
    for (int j = 0; j < 4; ++j) {
      v16bf b = fragB_f32(Br[j] + k, hf);
      acc[0][j] = wmma_bf16(a0, b, acc[0][j]);
      acc[1][j] = wmma_bf16(a1, b, acc[1][j]);
    }
  }
#pragma unroll
  for (int mi = 0; mi < 2; ++mi) {
#pragma unroll
    for (int j = 0; j < 4; ++j) {
      float bn = bo[n0 + j * 16 + n];
#pragma unroll
      for (int r = 0; r < 8; ++r) {
        size_t idx = (size_t)(m0 + mi * 16 + r + 8 * hf) * D_MODEL + n0 + j * 16 + n;
        y[idx] = acc[mi][j][r] + bn + hidden[idx];
      }
    }
  }
}

// ---------------- kernel 3b: LayerNorm (eps=1e-7) -----------------------------
__global__ void __launch_bounds__(256) ln_kernel(const float* __restrict__ y,
                                                 const float* __restrict__ g,
                                                 const float* __restrict__ be,
                                                 float* __restrict__ out) {
  __shared__ float red_s[8], red_q[8];
  int row = blockIdx.x;
  int tid = threadIdx.x;
  const float* x = y + (size_t)row * D_MODEL;
  float v0 = x[tid], v1 = x[tid + 256], v2 = x[tid + 512];
  float s = v0 + v1 + v2;
  float q = v0 * v0 + v1 * v1 + v2 * v2;
#pragma unroll
  for (int m = 16; m >= 1; m >>= 1) {
    s += __shfl_xor(s, m, 32);
    q += __shfl_xor(q, m, 32);
  }
  if ((tid & 31) == 0) { red_s[tid >> 5] = s; red_q[tid >> 5] = q; }
  __syncthreads();
  float ts = 0.f, tq = 0.f;
#pragma unroll
  for (int w = 0; w < 8; ++w) { ts += red_s[w]; tq += red_q[w]; }
  float mu = ts * (1.f / D_MODEL);
  float var = tq * (1.f / D_MODEL) - mu * mu;
  float rs = rsqrtf(var + 1e-7f);
  float* o = out + (size_t)row * D_MODEL;
  o[tid] = (v0 - mu) * rs * g[tid] + be[tid];
  o[tid + 256] = (v1 - mu) * rs * g[tid + 256] + be[tid + 256];
  o[tid + 512] = (v2 - mu) * rs * g[tid + 512] + be[tid + 512];
}

// ---------------- launch ------------------------------------------------------
extern "C" void kernel_launch(void* const* d_in, const int* in_sizes, int n_in,
                              void* d_out, int out_size, void* d_ws, size_t ws_size,
                              hipStream_t stream) {
  const float* hidden = (const float*)d_in[0];
  const float* rel = (const float*)d_in[1];
  const float* Wq = (const float*)d_in[2];
  const float* bq = (const float*)d_in[3];
  const float* Wk = (const float*)d_in[4];
  const float* bk = (const float*)d_in[5];
  const float* Wv = (const float*)d_in[6];
  const float* bv = (const float*)d_in[7];
  const float* Wo = (const float*)d_in[8];
  const float* bo = (const float*)d_in[9];
  const float* lng = (const float*)d_in[10];
  const float* lnb = (const float*)d_in[11];
  const int* amask = (const int*)d_in[12];
  (void)in_sizes; (void)n_in; (void)out_size; (void)ws_size;

  const size_t TOK = (size_t)BATCH * SEQ;          // 4096
  bf16_t* qb = (bf16_t*)d_ws;
  bf16_t* kb = qb + TOK * D_MODEL;
  bf16_t* vT = kb + TOK * D_MODEL;
  bf16_t* pqb = vT + TOK * D_MODEL;
  bf16_t* pkb = pqb + (size_t)2 * SPAN * D_MODEL;
  bf16_t* cx = pkb + (size_t)2 * SPAN * D_MODEL;
  float* yb = (float*)(cx + TOK * D_MODEL);

  // 32x64 tiles, 4 waves (128 threads) per block
  const int blkProj4096 = (int)((TOK / 32) * (D_MODEL / 64) / 4);      // 384
  const int blkProj1024 = (2 * SPAN / 32) * (D_MODEL / 64) / 4;        // 96

  proj_gemm_kernel<<<blkProj4096, 128, 0, stream>>>(hidden, Wq, bq, qb, (int)TOK, 0);
  proj_gemm_kernel<<<blkProj4096, 128, 0, stream>>>(hidden, Wk, bk, kb, (int)TOK, 0);
  proj_gemm_kernel<<<blkProj4096, 128, 0, stream>>>(hidden, Wv, bv, vT, (int)TOK, 1);
  proj_gemm_kernel<<<blkProj1024, 128, 0, stream>>>(rel, Wq, bq, pqb, 2 * SPAN, 0);
  proj_gemm_kernel<<<blkProj1024, 128, 0, stream>>>(rel, Wk, bk, pkb, 2 * SPAN, 0);

  const int attnBlocks = BATCH * NHEAD * (SEQ / 16) / 4;               // 768
  attn_kernel<<<attnBlocks, 128, 0, stream>>>(qb, kb, vT, pqb, pkb, amask, cx);

  out_gemm_kernel<<<blkProj4096, 128, 0, stream>>>(cx, Wo, bo, hidden, yb);
  ln_kernel<<<(int)TOK, 256, 0, stream>>>(yb, lng, lnb, (float*)d_out);
}
